// WeightDropLSTMCustom_47725676593290
// MI455X (gfx1250) — compile-verified, matrix operand order checked by reading, not run on hardware
//
#include <hip/hip_runtime.h>
#include <math.h>

// ---------------------------------------------------------------------------
// WeightDrop LSTM (T=512, B=64, I=H=1024) on MI455X / gfx1250 (wave32, WMMA)
//  Phase 1: gx[T*B,4H] = x @ Wx^T + bx + bh   -- bf16 WMMA, batch parallel
//  Phase 2: ONE persistent kernel, 64 blocks, loops 512 timesteps:
//           W_h tile staged in LDS once, B-operand from ds_load_b128,
//           c kept in registers for all steps, device-wide spin barrier.
// ---------------------------------------------------------------------------

typedef __attribute__((ext_vector_type(16))) __bf16 v16bf;
typedef __attribute__((ext_vector_type(8)))  __bf16 v8bf;
typedef __attribute__((ext_vector_type(8)))  float  v8f;

#define T_STEPS 512
#define BATCH   64
#define IDIM    1024
#define HDIM    1024
#define GDIM    (4 * HDIM)          // 4096
#define NROWS   (T_STEPS * BATCH)   // 32768
#define NBLK    64                  // persistent blocks (one h-col tile each)

// ---- fp32 -> bf16 round-to-nearest-even ----------------------------------
static __device__ __forceinline__ unsigned short f32_to_bf16(float f) {
    union { float f; unsigned u; } v; v.f = f;
    unsigned u = v.u;
    unsigned r = u + 0x7FFFu + ((u >> 16) & 1u);
    if ((u & 0x7F800000u) == 0x7F800000u) r = u;   // keep inf/nan
    return (unsigned short)(r >> 16);
}

__global__ void cvt_f32_bf16(const float* __restrict__ src,
                             unsigned short* __restrict__ dst, int n) {
    int i = blockIdx.x * blockDim.x + threadIdx.x;
    int stride = gridDim.x * blockDim.x;
    for (; i < n; i += stride) dst[i] = f32_to_bf16(src[i]);
}

__global__ void zero_u32(unsigned* p) { if (threadIdx.x == 0 && blockIdx.x == 0) *p = 0; }

// ---- 16x32 bf16 operand tile load (A or B), row-major, ld = 1024 ---------
// lanes 0-15: row = lane&15, K = {kb..kb+7, kb+16..kb+23};
// lanes 16-31: same rows, K offset +8 (kbase already includes it).
static __device__ __forceinline__ v16bf load_tile(const __bf16* __restrict__ base,
                                                  int row, int kbase) {
    const __bf16* p = base + (size_t)row * 1024 + kbase;
    const v8bf lo = *(const v8bf*)(p);
    const v8bf hi = *(const v8bf*)(p + 16);
    v16bf r;
#pragma unroll
    for (int i = 0; i < 8; ++i) { r[i] = lo[i]; r[i + 8] = hi[i]; }
    return r;
}

// Same pattern but from an LDS-resident [16][1024] bf16 tile -> ds_load_b128.
static __device__ __forceinline__ v16bf load_tile_lds(const unsigned short* base,
                                                      int row, int kbase) {
    const __bf16* p = (const __bf16*)base + row * 1024 + kbase;
    const v8bf lo = *(const v8bf*)(p);
    const v8bf hi = *(const v8bf*)(p + 16);
    v16bf r;
#pragma unroll
    for (int i = 0; i < 8; ++i) { r[i] = lo[i]; r[i + 8] = hi[i]; }
    return r;
}

// ---------------------------------------------------------------------------
// Phase 1: gx[row, g*H+j] = sum_k x[row,k] * Wx[g*H+j,k] + bx + bh
// One wave = 32 rows (M-unroll 2) x 16 cols x 4 gates; B reused across 2 rows.
// ---------------------------------------------------------------------------
__global__ __launch_bounds__(128)
void lstm_xproj(const unsigned short* __restrict__ xbf,   // [32768,1024] bf16
                const unsigned short* __restrict__ wxbf,  // [4096,1024]  bf16
                const float* __restrict__ bx,
                const float* __restrict__ bh,
                float* __restrict__ gx)                   // [32768,4096] f32
{
    const int lane     = threadIdx.x & 31;
    const int wid      = blockIdx.x * (blockDim.x >> 5) + (threadIdx.x >> 5);
    const int rowSuper = wid >> 6;           // 0..1023 (32 rows each)
    const int jTile    = wid & 63;           // 0..63
    const int rowBase  = rowSuper * 32;
    const int jBase    = jTile * 16;

    const __bf16* X = (const __bf16*)xbf;
    const __bf16* W = (const __bf16*)wxbf;

    const int laneLo  = lane & 15;
    const int laneHiK = (lane >> 4) * 8;

    v8f acc[2][4];
#pragma unroll
    for (int m = 0; m < 2; ++m)
#pragma unroll
        for (int g = 0; g < 4; ++g)
#pragma unroll
            for (int r = 0; r < 8; ++r) acc[m][g][r] = 0.0f;

    for (int k = 0; k < IDIM; k += 32) {
        const int kb = k + laneHiK;
        const v16bf a0 = load_tile(X, rowBase + laneLo, kb);
        const v16bf a1 = load_tile(X, rowBase + 16 + laneLo, kb);
#pragma unroll
        for (int g = 0; g < 4; ++g) {
            const v16bf b = load_tile(W, g * HDIM + jBase + laneLo, kb);
            acc[0][g] = __builtin_amdgcn_wmma_f32_16x16x32_bf16(
                false, a0, false, b, (short)0, acc[0][g], false, false);
            acc[1][g] = __builtin_amdgcn_wmma_f32_16x16x32_bf16(
                false, a1, false, b, (short)0, acc[1][g], false, false);
        }
    }

    // C/D layout: VGPR r -> M = r + 8*(lane>>4); N = lane&15
    const int n     = laneLo;
    const int mBase = (lane >> 4) * 8;
#pragma unroll
    for (int m = 0; m < 2; ++m) {
#pragma unroll
        for (int g = 0; g < 4; ++g) {
            const int col   = g * HDIM + jBase + n;
            const float bias = bx[col] + bh[col];
#pragma unroll
            for (int r = 0; r < 8; ++r) {
                const int row = rowBase + m * 16 + mBase + r;
                gx[(size_t)row * GDIM + col] = acc[m][g][r] + bias;
            }
        }
    }
}

// ---------------------------------------------------------------------------
// Phase 2: persistent recurrence. Grid = 64 blocks x 128 threads (4 waves).
// Block b owns h-columns [b*16, b*16+16); wave w = row-tile rt = w (0..3).
// W_h slice (4 gates x 16 rows x 1024 K bf16 = 128 KB, contiguous in global)
// staged to LDS once; c kept in registers across all 512 steps.
// ---------------------------------------------------------------------------
__global__ __launch_bounds__(128)
void lstm_recurrent(const float* __restrict__ gx,          // [T*B,4096]
                    const unsigned short* __restrict__ whbf,// [4096,1024] bf16
                    unsigned short* __restrict__ hb0,      // [64,1024] bf16
                    unsigned short* __restrict__ hb1,      // [64,1024] bf16
                    const float* __restrict__ c0,          // [64,1024]
                    float* __restrict__ out,               // [T,B,H]++[B,H]++[B,H]
                    unsigned* __restrict__ bar)
{
    __shared__ unsigned short sW[4][16 * 1024];            // 128 KB

    const int tid     = threadIdx.x;
    const int lane    = tid & 31;
    const int rt      = tid >> 5;          // 0..3  row tile
    const int jBase   = blockIdx.x * 16;
    const int rowBase = rt * 16;
    const int laneLo  = lane & 15;
    const int laneHiK = (lane >> 4) * 8;
    const int mBase   = (lane >> 4) * 8;
    const int n       = laneLo;
    const int hcol    = jBase + n;

    // ---- stage W_h slice into LDS (4 contiguous 32KB chunks) -------------
    {
#pragma unroll
        for (int idx = tid; idx < 8192; idx += 128) {      // 8192 x 16B = 128KB
            const int g  = idx >> 11;                      // 2048 uint4 / chunk
            const int ic = idx & 2047;
            const uint4* __restrict__ src =
                (const uint4*)(whbf + (size_t)(g * HDIM + jBase) * 1024);
            ((uint4*)&sW[g][0])[ic] = src[ic];
        }
    }
    __syncthreads();

    // ---- cell state lives in registers for the whole sequence -----------
    v8f creg, hreg;
#pragma unroll
    for (int r = 0; r < 8; ++r) {
        const int b = rowBase + mBase + r;
        creg[r] = c0[(size_t)b * HDIM + hcol];
        hreg[r] = 0.0f;
    }

    for (int t = 0; t < T_STEPS; ++t) {
        const unsigned short* hin  = (t & 1) ? hb1 : hb0;
        unsigned short*       hout = (t & 1) ? hb0 : hb1;
        const float* gx_t = gx + (size_t)t * BATCH * GDIM;
        const __bf16* H = (const __bf16*)hin;
        float* out_t = out + (size_t)t * BATCH * HDIM;

        v8f acc[4];
#pragma unroll
        for (int g = 0; g < 4; ++g) {
            const int col = g * HDIM + jBase + n;
#pragma unroll
            for (int r = 0; r < 8; ++r)
                acc[g][r] = gx_t[(size_t)(rowBase + mBase + r) * GDIM + col];
        }

        for (int k = 0; k < HDIM; k += 32) {
            const int kb = k + laneHiK;
            const v16bf a = load_tile(H, rowBase + laneLo, kb);
#pragma unroll
            for (int g = 0; g < 4; ++g) {
                const v16bf b = load_tile_lds(&sW[g][0], laneLo, kb);
                acc[g] = __builtin_amdgcn_wmma_f32_16x16x32_bf16(
                    false, a, false, b, (short)0, acc[g], false, false);
            }
        }

#pragma unroll
        for (int r = 0; r < 8; ++r) {
            const int b = rowBase + mBase + r;
            const float xi = acc[0][r], xf = acc[1][r];
            const float xo = acc[2][r], xn = acc[3][r];
            const float ig = 1.0f / (1.0f + __expf(-xi));
            const float fg = 1.0f / (1.0f + __expf(-xf));
            const float og = 1.0f / (1.0f + __expf(-xo));
            const float ng = tanhf(xn);
            const float c_new = fg * creg[r] + ig * ng;
            const float h_new = og * tanhf(c_new);
            creg[r] = c_new;
            hreg[r] = h_new;
            const size_t idx = (size_t)b * HDIM + hcol;
            out_t[idx]  = h_new;
            hout[idx]   = f32_to_bf16(h_new);
        }

        // ---- device-wide barrier: all h writes visible before next read --
        __threadfence();
        __syncthreads();
        if (tid == 0) {
            __hip_atomic_fetch_add(bar, 1u, __ATOMIC_ACQ_REL,
                                   __HIP_MEMORY_SCOPE_AGENT);
            const unsigned target = (unsigned)(t + 1) * NBLK;
            while (__hip_atomic_load(bar, __ATOMIC_ACQUIRE,
                                     __HIP_MEMORY_SCOPE_AGENT) < target)
                __builtin_amdgcn_s_sleep(1);
        }
        __syncthreads();
        __threadfence();
    }

    // ---- finals from registers: h_fin = out[T-1] values, c_fin = creg ----
    float* h_fin = out + (size_t)T_STEPS * BATCH * HDIM;
    float* c_fin = h_fin + (size_t)BATCH * HDIM;
#pragma unroll
    for (int r = 0; r < 8; ++r) {
        const int b = rowBase + mBase + r;
        const size_t idx = (size_t)b * HDIM + hcol;
        h_fin[idx] = hreg[r];
        c_fin[idx] = creg[r];
    }
}

// ---------------------------------------------------------------------------
extern "C" void kernel_launch(void* const* d_in, const int* in_sizes, int n_in,
                              void* d_out, int out_size, void* d_ws, size_t ws_size,
                              hipStream_t stream) {
    const float* x  = (const float*)d_in[0];   // [T,B,I]
    const float* h0 = (const float*)d_in[1];   // [B,H]
    const float* c0 = (const float*)d_in[2];   // [B,H]
    const float* Wx = (const float*)d_in[3];   // [4H,I]
    const float* Wh = (const float*)d_in[4];   // [4H,H]
    const float* bx = (const float*)d_in[5];   // [4H]
    const float* bh = (const float*)d_in[6];   // [4H]
    float* out = (float*)d_out;

    char* ws = (char*)d_ws;
    size_t off = 0;
    auto carve = [&](size_t bytes) -> void* {
        void* p = (void*)(ws + off);
        off += (bytes + 255) & ~(size_t)255;
        return p;
    };
    unsigned short* xbf  = (unsigned short*)carve((size_t)NROWS * IDIM * 2); // 64 MB
    unsigned short* wxbf = (unsigned short*)carve((size_t)GDIM * IDIM * 2);  // 8 MB
    unsigned short* whbf = (unsigned short*)carve((size_t)GDIM * HDIM * 2);  // 8 MB
    unsigned short* hb0  = (unsigned short*)carve((size_t)BATCH * HDIM * 2);
    unsigned short* hb1  = (unsigned short*)carve((size_t)BATCH * HDIM * 2);
    unsigned*       bar  = (unsigned*)carve(256);
    float*          gx   = (float*)carve((size_t)NROWS * GDIM * 4);          // 512 MB

    // ---- Phase 0: bf16 staging + barrier init -----------------------------
    cvt_f32_bf16<<<4096, 256, 0, stream>>>(x,  xbf,  NROWS * IDIM);
    cvt_f32_bf16<<<2048, 256, 0, stream>>>(Wx, wxbf, GDIM * IDIM);
    cvt_f32_bf16<<<2048, 256, 0, stream>>>(Wh, whbf, GDIM * HDIM);
    cvt_f32_bf16<<<256,  256, 0, stream>>>(h0, hb0,  BATCH * HDIM);
    zero_u32<<<1, 32, 0, stream>>>(bar);

    // ---- Phase 1: input-projection GEMM (275 GFLOP, batch parallel) -------
    // 65536 waves (1024 row-supertiles x 64 h-tiles), 4 waves / block.
    lstm_xproj<<<16384, 128, 0, stream>>>(xbf, wxbf, bx, bh, gx);

    // ---- Phase 2: single persistent kernel runs all 512 timesteps ---------
    lstm_recurrent<<<NBLK, 128, 0, stream>>>(gx, whbf, hb0, hb1, c0, out, bar);
}
